// DirectVoxGO_33603824124161
// MI455X (gfx1250) — matrix operand order repaired and checked by reading, not compiled, not run on hardware
//
#include <hip/hip_runtime.h>
#include <hip/hip_bf16.h>
#include <math.h>

typedef __attribute__((ext_vector_type(16))) _Float16 v16h;
typedef __attribute__((ext_vector_type(8)))  _Float16 v8h;
typedef __attribute__((ext_vector_type(8)))  float    v8f;

#define G      160
#define G3     (160*160*160)
#define NS     256          // samples per ray
#define WIDTH  128          // hidden width
#define K0C    12           // feature-grid channels
#define KPAD   64           // padded input dim (39 -> 64)
#define ACT_SHIFT_C (-4.59511985013459f)
#define INTERVAL_C  0.5f

// Assemble a 16-half fragment from two contiguous 8-half (16-byte) LDS loads.
__device__ __forceinline__ v16h load_contig16(const _Float16* p) {
    const v8h lo = *(const v8h*)(p);
    const v8h hi = *(const v8h*)(p + 8);
    v16h r;
#pragma unroll
    for (int e = 0; e < 8; ++e) { r[e] = lo[e]; r[e + 8] = hi[e]; }
    return r;
}

// A-fragment (16x32 f16, wave32): lane L<16 -> row M=L, halves cover
// K = cbase + {0..7} and cbase + {16..23}; lanes 16..31 same rows, K offset +8.
__device__ __forceinline__ v16h load_a_frag(const _Float16* row, int cbase, int khiA) {
    const v8h lo = *(const v8h*)(row + cbase + khiA);
    const v8h hi = *(const v8h*)(row + cbase + 16 + khiA);
    v16h a;
#pragma unroll
    for (int e = 0; e < 8; ++e) { a[e] = lo[e]; a[e + 8] = hi[e]; }
    return a;
}

__global__ __launch_bounds__(256, 1)
void dvgo_fused_kernel(const float* __restrict__ pts, const float* __restrict__ viewdirs,
                       const float* __restrict__ dgrid, const float* __restrict__ kgrid,
                       const float* __restrict__ w0, const float* __restrict__ b0,
                       const float* __restrict__ w1, const float* __restrict__ b1,
                       const float* __restrict__ w2, const float* __restrict__ b2,
                       float* __restrict__ out)
{
    // ---- LDS (~122 KB; CDNA5 allows up to 320 KB/workgroup) ----
    __shared__ __align__(16) _Float16 featS[NS][KPAD];       // 32 KB  per-sample features (f16, padded)
    __shared__ __align__(16) _Float16 w0T[WIDTH][KPAD];      // 16 KB  w0^T padded K 39->64
    __shared__ __align__(16) _Float16 w1T[WIDTH][WIDTH];     // 32 KB  w1^T
    __shared__ __align__(16) _Float16 w2T[16][WIDTH];        //  4 KB  w2^T padded N 3->16
    __shared__ __align__(16) _Float16 hst[8][16][WIDTH];     // 32 KB  per-wave activation staging
    __shared__ float b0S[WIDTH], b1S[WIDTH], b2S[16];
    __shared__ __align__(16) _Float16 embf[KPAD];            // view embedding in feat columns 12..38
    __shared__ float alphaS[NS], wgtS[NS];
    __shared__ float rgbsum[3];
    __shared__ float ainvS;

    const int tid  = threadIdx.x;
    const int ray  = blockIdx.x;
    const int lane = tid & 31;
    const int wv   = tid >> 5;              // wave id 0..7

    // ================= Phase A: stage weights (transposed f16) + ray embedding =================
    for (int i = tid; i < WIDTH * KPAD; i += 256) {
        int n = i >> 6, k = i & 63;
        w0T[n][k] = (k < 39) ? (_Float16)w0[k * WIDTH + n] : (_Float16)0.f;
    }
    for (int i = tid; i < WIDTH * WIDTH; i += 256) {
        int n = i >> 7, k = i & 127;
        w1T[n][k] = (_Float16)w1[k * WIDTH + n];
    }
    for (int i = tid; i < 16 * WIDTH; i += 256) {
        int n = i >> 7, k = i & 127;
        w2T[n][k] = (n < 3) ? (_Float16)w2[k * 3 + n] : (_Float16)0.f;
    }
    if (tid < WIDTH) { b0S[tid] = b0[tid]; b1S[tid] = b1[tid]; }
    if (tid < 16) b2S[tid] = (tid < 3) ? b2[tid] : 0.f;
    if (tid < 3)  rgbsum[tid] = 0.f;
    if (tid == 0) {
        float vx = viewdirs[ray*3+0], vy = viewdirs[ray*3+1], vz = viewdirs[ray*3+2];
        float inv = rsqrtf(vx*vx + vy*vy + vz*vz);
        float vd[3] = {vx*inv, vy*inv, vz*inv};
#pragma unroll
        for (int k = 0; k < KPAD; ++k) embf[k] = (_Float16)0.f;
        embf[12] = (_Float16)vd[0]; embf[13] = (_Float16)vd[1]; embf[14] = (_Float16)vd[2];
#pragma unroll
        for (int d = 0; d < 3; ++d)
#pragma unroll
            for (int j = 0; j < 4; ++j) {
                float ang = vd[d] * (float)(1 << j);
                embf[15 + d*4 + j] = (_Float16)__sinf(ang);
                embf[27 + d*4 + j] = (_Float16)__cosf(ang);
            }
    }
    __syncthreads();

    // ================= Phase B: trilinear gather (one sample per thread) =================
    {
        const int s = tid;
        const float* p = pts + ((size_t)ray * NS + s) * 3;
        int i0[3]; float f[3];
#pragma unroll
        for (int d = 0; d < 3; ++d) {
            float uu = (p[d] + 1.f) * 0.5f * (float)(G - 1);
            uu = fminf(fmaxf(uu, 0.f), (float)(G - 1));
            int ii = (int)uu;
            ii = (ii > G - 2) ? (G - 2) : ii;
            i0[d] = ii; f[d] = uu - (float)ii;
        }
        const float fx = f[0], fy = f[1], fz = f[2];
        const int idx000 = (i0[0]*G + i0[1])*G + i0[2];
        int idx[8];
        idx[0]=idx000;         idx[1]=idx000+1;
        idx[2]=idx000+G;       idx[3]=idx000+G+1;
        idx[4]=idx000+G*G;     idx[5]=idx000+G*G+1;
        idx[6]=idx000+G*G+G;   idx[7]=idx000+G*G+G+1;
        float cw[8];
        cw[0]=(1-fx)*(1-fy)*(1-fz); cw[1]=(1-fx)*(1-fy)*fz;
        cw[2]=(1-fx)*fy*(1-fz);     cw[3]=(1-fx)*fy*fz;
        cw[4]=fx*(1-fy)*(1-fz);     cw[5]=fx*(1-fy)*fz;
        cw[6]=fx*fy*(1-fz);         cw[7]=fx*fy*fz;

        float dens = 0.f;
#pragma unroll
        for (int c8 = 0; c8 < 8; ++c8) dens += cw[c8] * dgrid[idx[c8]];
        float x  = dens + ACT_SHIFT_C;
        float sp = (x > 20.f) ? x : log1pf(__expf(x));
        alphaS[s] = 1.f - __expf(-sp * INTERVAL_C);

#pragma unroll
        for (int c = 0; c < K0C; ++c) {
            const float* gc = kgrid + (size_t)c * G3;
            float v = 0.f;
#pragma unroll
            for (int c8 = 0; c8 < 8; ++c8) v += cw[c8] * gc[idx[c8]];
            featS[s][c] = (_Float16)v;
        }
        for (int k = K0C; k < KPAD; ++k) featS[s][k] = embf[k];
    }
    __syncthreads();

    // ================= Phase C: transmittance scan =================
    if (tid == 0) {
        float T = 1.f;
        for (int s = 0; s < NS; ++s) {
            float a = alphaS[s];
            wgtS[s] = a * T;
            T *= (1.f - a);
        }
        ainvS = T;
    }
    __syncthreads();

    // ================= Phase D: WMMA MLP, two 16-sample tiles per wave =================
    const int row16 = lane & 15;
    const int hi    = lane >> 4;        // K-half select
    const int khiA  = hi * 8;           // A fragment +8 K for high lanes
    const int koffB = hi * 16;          // unused: B fragments also use khiA (+8 K)

    for (int tt = 0; tt < 2; ++tt) {
        const int tile = wv + tt * 8;
        const int s0   = tile * 16;
        const _Float16* arow = &featS[s0 + row16][0];
        const v16h a0 = load_a_frag(arow, 0,  khiA);
        const v16h a1 = load_a_frag(arow, 32, khiA);

        // ---- layer 0: [16x64] @ [64x128] ----
#pragma unroll
        for (int n = 0; n < 8; ++n) {
            v8f acc = {};
            const _Float16* bcol = &w0T[n*16 + row16][0];
            const v16h bf0 = load_a_frag(bcol, 0,  khiA);
            const v16h bf1 = load_a_frag(bcol, 32, khiA);
            acc = __builtin_amdgcn_wmma_f32_16x16x32_f16(false, a0, false, bf0, (short)0, acc, false, false);
            acc = __builtin_amdgcn_wmma_f32_16x16x32_f16(false, a1, false, bf1, (short)0, acc, false, false);
            const int   nc = n*16 + row16;
            const float bb = b0S[nc];
#pragma unroll
            for (int r = 0; r < 8; ++r) {
                float v = fmaxf(acc[r] + bb, 0.f);
                hst[wv][r + 8*hi][nc] = (_Float16)v;
            }
        }
        __builtin_amdgcn_wave_barrier();

        // ---- layer 1: [16x128] @ [128x128] ----
        const _Float16* hrow = &hst[wv][row16][0];
        v16h ha[4];
#pragma unroll
        for (int c = 0; c < 4; ++c) ha[c] = load_a_frag(hrow, c*32, khiA);
        __builtin_amdgcn_wave_barrier();
#pragma unroll
        for (int n = 0; n < 8; ++n) {
            v8f acc = {};
            const _Float16* bcol = &w1T[n*16 + row16][0];
#pragma unroll
            for (int c = 0; c < 4; ++c) {
                const v16h bf = load_a_frag(bcol, c*32, khiA);
                acc = __builtin_amdgcn_wmma_f32_16x16x32_f16(false, ha[c], false, bf, (short)0, acc, false, false);
            }
            const int   nc = n*16 + row16;
            const float bb = b1S[nc];
#pragma unroll
            for (int r = 0; r < 8; ++r) {
                float v = fmaxf(acc[r] + bb, 0.f);
                hst[wv][r + 8*hi][nc] = (_Float16)v;
            }
        }
        __builtin_amdgcn_wave_barrier();

        // ---- layer 2: [16x128] @ [128x16(pad of 3)] + sigmoid + weighted reduce ----
        v16h ga[4];
#pragma unroll
        for (int c = 0; c < 4; ++c) ga[c] = load_a_frag(hrow, c*32, khiA);
        v8f acc2 = {};
#pragma unroll
        for (int c = 0; c < 4; ++c) {
            const _Float16* bcol = &w2T[row16][0];
            const v16h bf = load_a_frag(bcol, c*32, khiA);
            acc2 = __builtin_amdgcn_wmma_f32_16x16x32_f16(false, ga[c], false, bf, (short)0, acc2, false, false);
        }
        // C layout: lane holds N = row16 for M = r + 8*hi -> sample s0 + M
        if (row16 < 3) {
            const float bb = b2S[row16];
#pragma unroll
            for (int r = 0; r < 8; ++r) {
                float v  = acc2[r] + bb;
                float sg = 1.f / (1.f + __expf(-v));
                const int s = s0 + r + 8*hi;
                atomicAdd(&rgbsum[row16], wgtS[s] * sg);
            }
        }
    }
    __syncthreads();

    // ================= Phase E: output =================
    if (tid < 3) out[(size_t)ray * 3 + tid] = rgbsum[tid] + ainvS;
}

extern "C" void kernel_launch(void* const* d_in, const int* in_sizes, int n_in,
                              void* d_out, int out_size, void* d_ws, size_t ws_size,
                              hipStream_t stream) {
    const float* pts      = (const float*)d_in[0];
    const float* viewdirs = (const float*)d_in[1];
    const float* dgrid    = (const float*)d_in[2];
    const float* kgrid    = (const float*)d_in[3];
    const float* w0 = (const float*)d_in[4];
    const float* b0 = (const float*)d_in[5];
    const float* w1 = (const float*)d_in[6];
    const float* b1 = (const float*)d_in[7];
    const float* w2 = (const float*)d_in[8];
    const float* b2 = (const float*)d_in[9];
    float* out = (float*)d_out;

    const int N = in_sizes[1] / 3;   // rays (viewdirs is [N,3]); S fixed at 256
    dvgo_fused_kernel<<<N, 256, 0, stream>>>(pts, viewdirs, dgrid, kgrid,
                                             w0, b0, w1, b1, w2, b2, out);
    (void)d_ws; (void)ws_size; (void)n_in; (void)out_size;
}
